// L2Conv2D_56470230008360
// MI455X (gfx1250) — compile-verified, matrix operand order checked by reading, not run on hardware
//
#include <hip/hip_runtime.h>
#include <math.h>

// ---------------- types ----------------
typedef __bf16 bf16;
typedef bf16  v16bf __attribute__((ext_vector_type(16)));
typedef bf16  v8bf  __attribute__((ext_vector_type(8)));
typedef float v8f   __attribute__((ext_vector_type(8)));

// ---------------- problem constants ----------------
#define NB   32
#define CIN  256
#define WIN  28
#define NP   1024
#define OW   26
#define KTOT 2304        // 256*9
#define MTOT 21632       // 32*26*26 = 169*128
#define PLANE 784
#define IMG   200704
#define OPLANE 676

#define BM 256           // p-tile   (1024 = 4*256)
#define BN 128           // m-tile   (21632 = 169*128)
#define BK 32            // K chunk == one WMMA K
#define BKP 40           // padded LDS row (80B) -> conflict-free b128 frag reads

// =====================================================================
// ||proto||^2 per prototype -> psq[p]
// =====================================================================
__global__ __launch_bounds__(256) void psq_kernel(const float* __restrict__ proto,
                                                  float* __restrict__ psq) {
  int p = blockIdx.x;
  const float* row = proto + (size_t)p * KTOT;
  float acc = 0.f;
  for (int k = threadIdx.x; k < KTOT; k += 256) {
    float v = row[k];
    acc += v * v;
  }
  for (int off = 16; off > 0; off >>= 1) acc += __shfl_down(acc, off, 32);
  __shared__ float wsum[8];
  if ((threadIdx.x & 31) == 0) wsum[threadIdx.x >> 5] = acc;
  __syncthreads();
  if (threadIdx.x == 0) {
    float t = 0.f;
    #pragma unroll
    for (int i = 0; i < 8; ++i) t += wsum[i];
    psq[p] = t;
  }
}

// =====================================================================
// ||patch||^2 per output position -> xsq[m]
// =====================================================================
__global__ __launch_bounds__(128) void xsq_kernel(const float* __restrict__ x,
                                                  float* __restrict__ xsq) {
  int m = blockIdx.x;
  int nimg = m / OPLANE;
  int s = m - nimg * OPLANE;
  int oh = s / OW;
  int ow = s - oh * OW;
  const float* xb = x + (size_t)nimg * IMG + oh * WIN + ow;
  float acc = 0.f;
  for (int k = threadIdx.x; k < KTOT; k += 128) {
    int c  = k / 9;
    int r  = k - 9 * c;
    int d3 = (r * 11) >> 5;            // r/3 for r<9
    float v = xb[c * PLANE + d3 * WIN + (r - 3 * d3)];
    acc += v * v;
  }
  for (int off = 16; off > 0; off >>= 1) acc += __shfl_down(acc, off, 32);
  __shared__ float wsum[4];
  if ((threadIdx.x & 31) == 0) wsum[threadIdx.x >> 5] = acc;
  __syncthreads();
  if (threadIdx.x == 0) xsq[m] = wsum[0] + wsum[1] + wsum[2] + wsum[3];
}

// =====================================================================
// Implicit-GEMM L2 distance (bf16x3 split, v_wmma_f32_16x16x32_bf16)
// Workgroup tile 256(p) x 128(m); 8 waves of 64x64; K-chunk 32.
// Register-level software pipeline on global loads; padded LDS rows.
// =====================================================================
__device__ inline v16bf load_frag(const bf16 (*tile)[BKP], int row, int h) {
  const v8bf* rp = (const v8bf*)(&tile[row][0]);   // 80B row, 16B groups
  v8bf a = rp[h];          // K 0-7  (h=0) / K 8-15  (h=1)
  v8bf b = rp[2 + h];      // K16-23 (h=0) / K 24-31 (h=1)
  return __builtin_shufflevector(a, b, 0, 1, 2, 3, 4, 5, 6, 7,
                                       8, 9, 10, 11, 12, 13, 14, 15);
}

__device__ inline void split_bf16(float f, bf16& hi, bf16& lo) {
  bf16 h = (bf16)f;
  hi = h;
  lo = (bf16)(f - (float)h);
}

// issue raw f32 global loads for one K-chunk into registers
__device__ inline void gload(const float* __restrict__ aprow,
                             const float* __restrict__ xb, int bhalf, int kc,
                             float av[32], float bv[16]) {
  const float4* ap = (const float4*)(aprow + kc);
  #pragma unroll
  for (int q = 0; q < 8; ++q) *(float4*)&av[q * 4] = ap[q];
  int k0 = kc + bhalf;
  int c  = k0 / 9;
  int r  = k0 - 9 * c;
  #pragma unroll
  for (int i = 0; i < 16; ++i) {
    int d3 = (r * 11) >> 5;                        // r/3 for r<9
    bv[i] = xb[c * PLANE + d3 * WIN + (r - 3 * d3)];
    ++r;
    if (r == 9) { r = 0; ++c; }
  }
}

__global__ __launch_bounds__(256) void l2conv_wmma_kernel(
    const float* __restrict__ x, const float* __restrict__ proto,
    const float* __restrict__ xsq, const float* __restrict__ psq,
    float* __restrict__ out) {
  __shared__ __align__(16) bf16 Ah[BM][BKP];   // proto hi (256 rows, 80B each)
  __shared__ __align__(16) bf16 Al[BM][BKP];   // proto lo
  __shared__ __align__(16) bf16 Bh[BN][BKP];   // patch hi ([m_local][k])
  __shared__ __align__(16) bf16 Bl[BN][BKP];   // patch lo

  const int m0 = blockIdx.x * BN;   // patch-tile base  (169 blocks)
  const int p0 = blockIdx.y * BM;   // proto-tile base  (4 blocks)

  const int tid  = threadIdx.x;
  const int lane = tid & 31;
  const int wid  = tid >> 5;        // 0..7
  const int l    = lane & 15;
  const int h    = lane >> 4;
  const int wp   = wid >> 1;        // 0..3 : 64-row p slice
  const int wm   = wid & 1;         // 0..1 : 64-col m slice
  const int pbase = wp * 64;
  const int mbase = wm * 64;

  // ---- A loader: one full 32-k row per thread ----
  const float* aprow = proto + (size_t)(p0 + tid) * KTOT;

  // ---- B loader: 2 threads per patch row, 16 gathered k each ----
  const int brow  = tid >> 1;
  const int bhalf = (tid & 1) << 4;
  const int bm    = m0 + brow;
  const int bnimg = bm / OPLANE;
  const int bs    = bm - bnimg * OPLANE;
  const int boh   = bs / OW;
  const int bow   = bs - boh * OW;
  const float* xb = x + (size_t)bnimg * IMG + boh * WIN + bow;

  v8f acc[4][4];
  #pragma unroll
  for (int tp = 0; tp < 4; ++tp)
    #pragma unroll
    for (int tm = 0; tm < 4; ++tm)
      acc[tp][tm] = (v8f)(0.f);

  // ---- software pipeline: prefetch chunk 0 ----
  float av[32], bv[16];
  gload(aprow, xb, bhalf, 0, av, bv);

  for (int kc = 0; kc < KTOT; kc += BK) {
    __syncthreads();   // previous compute done before overwriting LDS

    // ---- stage current chunk: f32 -> (hi,lo) bf16 -> LDS ----
    #pragma unroll
    for (int g = 0; g < 4; ++g) {
      v8bf hh, ll;
      #pragma unroll
      for (int i = 0; i < 8; ++i) {
        bf16 hi, lo;
        split_bf16(av[g * 8 + i], hi, lo);
        hh[i] = hi; ll[i] = lo;
      }
      *(v8bf*)&Ah[tid][g * 8] = hh;
      *(v8bf*)&Al[tid][g * 8] = ll;
    }
    #pragma unroll
    for (int g = 0; g < 2; ++g) {
      v8bf hh, ll;
      #pragma unroll
      for (int i = 0; i < 8; ++i) {
        bf16 hi, lo;
        split_bf16(bv[g * 8 + i], hi, lo);
        hh[i] = hi; ll[i] = lo;
      }
      *(v8bf*)&Bh[brow][bhalf + g * 8] = hh;
      *(v8bf*)&Bl[brow][bhalf + g * 8] = ll;
    }

    __syncthreads();

    // ---- prefetch next chunk (loads in flight across the WMMA block) ----
    if (kc + BK < KTOT) gload(aprow, xb, bhalf, kc + BK, av, bv);

    // ---- fragments: 4 A-frags + 4 B-frags (hi+lo), conflict-free b128 ----
    v16bf ah[4], al[4], bh[4], bl[4];
    #pragma unroll
    for (int tp = 0; tp < 4; ++tp) {
      ah[tp] = load_frag(Ah, pbase + tp * 16 + l, h);
      al[tp] = load_frag(Al, pbase + tp * 16 + l, h);
    }
    #pragma unroll
    for (int tm = 0; tm < 4; ++tm) {
      bh[tm] = load_frag(Bh, mbase + tm * 16 + l, h);
      bl[tm] = load_frag(Bl, mbase + tm * 16 + l, h);
    }

    // ---- bf16x3 WMMA: hi*hi + hi*lo + lo*hi (48 WMMAs / iter / wave) ----
    #pragma unroll
    for (int tp = 0; tp < 4; ++tp)
      #pragma unroll
      for (int tm = 0; tm < 4; ++tm) {
        acc[tp][tm] = __builtin_amdgcn_wmma_f32_16x16x32_bf16(
            false, ah[tp], false, bh[tm], (short)0, acc[tp][tm], false, false);
        acc[tp][tm] = __builtin_amdgcn_wmma_f32_16x16x32_bf16(
            false, ah[tp], false, bl[tm], (short)0, acc[tp][tm], false, false);
        acc[tp][tm] = __builtin_amdgcn_wmma_f32_16x16x32_bf16(
            false, al[tp], false, bh[tm], (short)0, acc[tp][tm], false, false);
      }
  }

  // ---- epilogue: d2 = xsq - 2*cross + psq ; out = sqrt(clamp) ----
  #pragma unroll
  for (int tm = 0; tm < 4; ++tm) {
    int m    = m0 + mbase + tm * 16 + l;     // this lane's patch column
    int nimg = m / OPLANE;
    int s    = m - nimg * OPLANE;
    float xs = xsq[m];
    float* op = out + (size_t)nimg * (size_t)NP * OPLANE + s;
    #pragma unroll
    for (int tp = 0; tp < 4; ++tp) {
      #pragma unroll
      for (int r = 0; r < 8; ++r) {
        int p = p0 + pbase + tp * 16 + r + 8 * h;   // D-matrix row layout
        float cr = acc[tp][tm][r];
        float d2 = xs - 2.0f * cr + psq[p];
        op[(size_t)p * OPLANE] = sqrtf(fmaxf(d2, 1e-14f));
      }
    }
  }
}

// =====================================================================
extern "C" void kernel_launch(void* const* d_in, const int* in_sizes, int n_in,
                              void* d_out, int out_size, void* d_ws, size_t ws_size,
                              hipStream_t stream) {
  const float* x     = (const float*)d_in[0];   // (32,256,28,28)
  const float* proto = (const float*)d_in[1];   // (1024,256,3,3)
  float* out = (float*)d_out;                   // (32,1024,26,26)

  float* xsq = (float*)d_ws;                    // 21632 f32
  float* psq = xsq + MTOT;                      // 1024 f32

  psq_kernel<<<NP, 256, 0, stream>>>(proto, psq);
  xsq_kernel<<<MTOT, 128, 0, stream>>>(x, xsq);

  dim3 grid(MTOT / BN, NP / BM);                // 169 x 4
  l2conv_wmma_kernel<<<grid, 256, 0, stream>>>(x, proto, xsq, psq, out);
}
